// StockPricePredictor1_49246095015972
// MI455X (gfx1250) — matrix operand order repaired and checked by reading, not compiled, not run on hardware
//
#include <hip/hip_runtime.h>

// ---------------- types ----------------
typedef __attribute__((ext_vector_type(8)))  float  v8f;
typedef __attribute__((ext_vector_type(8)))  __bf16 v8bf;
typedef __attribute__((ext_vector_type(16))) __bf16 v16bf;

#define SEQ_LEN    4096
#define INPUT_SIZE 128
#define HIDDEN     2048
#define NGATE      (4 * HIDDEN)      // 8192 gate rows (i,f,g,o)
#define NWG        64
#define WGSIZE     256
#define KTILES     (HIDDEN / 32)     // 64 WMMA K-steps per matvec
#define NTILES     (NGATE / 16)      // 512 N-tiles == total waves

__device__ __forceinline__ __bf16 f2bf(float f) {
  union { float f; unsigned u; } v; v.f = f;
  unsigned r = (v.u + 0x7fffu + ((v.u >> 16) & 1u)) >> 16;  // RNE
  unsigned short s = (unsigned short)r;
  __bf16 b; __builtin_memcpy(&b, &s, 2);
  return b;
}

// ---------------- kernel 1: reset grid-barrier counter ----------------
__global__ void init_ws(unsigned* bar) {
  if (threadIdx.x == 0) *bar = 0u;
}

// ---------------- kernel 2: W_hh (fp32, row-major [8192,2048]) ->
// bf16 fragment-linear layout for the 16x16x32 bf16 B operand, k contiguous per lane half (same order as A).
// dst[((ntile*64 + ktile)*32 + lane)*16 + e]:
//   lane<16 : column N = ntile*16+lane,    K = ktile*32 + e       (e=0..15)
//   lane>=16: column N = ntile*16+lane-16, K = ktile*32 + 16 + e
__global__ void swizzle_whh(const float* __restrict__ whh, __bf16* __restrict__ wsw) {
  unsigned d = blockIdx.x * 256u + threadIdx.x;   // < 8192*2048 = 16.7M
  unsigned e    = d & 15u;
  unsigned lane = (d >> 4) & 31u;
  unsigned kt   = (d >> 9) & 63u;
  unsigned nt   = d >> 15;                        // 0..511
  unsigned n = nt * 16u + (lane & 15u);
  unsigned k = kt * 32u + (lane & 16u) + e;
  wsw[d] = f2bf(whh[(size_t)n * HIDDEN + k]);
}

// ---------------- kernel 3: gx[t][n] = x[t] . W_ih[n] + b_ih[n] + b_hh[n]
// block = (t_tile of 32, n_tile of 256); each thread owns one n, 32 t's.
__global__ void gx_kernel(const float* __restrict__ x, const float* __restrict__ wih,
                          const float* __restrict__ bih, const float* __restrict__ bhh,
                          float* __restrict__ gx) {
  __shared__ float xt[32 * INPUT_SIZE];           // 16 KB input tile
  unsigned bid = blockIdx.x;                      // 0..4095
  unsigned tt  = bid >> 5;                        // time tile (32 steps)
  unsigned nt  = bid & 31u;                       // n tile (256 rows)
  unsigned tid = threadIdx.x;
  #pragma unroll
  for (int r = 0; r < 16; ++r)
    xt[r * 256 + tid] = x[(size_t)tt * 32 * INPUT_SIZE + r * 256 + tid];
  __syncthreads();
  unsigned n = nt * 256u + tid;
  float b = bih[n] + bhh[n];
  float acc[32];
  #pragma unroll
  for (int t = 0; t < 32; ++t) acc[t] = b;
  const float* wr = wih + (size_t)n * INPUT_SIZE;
  for (int k = 0; k < INPUT_SIZE; ++k) {
    float w = wr[k];
    #pragma unroll
    for (int t = 0; t < 32; ++t) acc[t] += w * xt[t * INPUT_SIZE + k];
  }
  for (int t = 0; t < 32; ++t)
    gx[(size_t)(tt * 32 + t) * NGATE + n] = acc[t];
}

// ---------------- kernel 4: persistent LSTM recurrence ----------------
// 64 WGs x 256 threads = 512 waves; wave g owns gate rows [16g, 16g+16).
// Per step: 64x v_wmma_f32_16x16x32_bf16 accumulating h . W_hh.T for its tile,
// + gx, store to double-buffered global gates, grid barrier, then every WG
// redundantly does the i/f/g/o elementwise update (c in registers, h -> LDS bf16).
__launch_bounds__(WGSIZE, 1)
__global__ void lstm_persistent(const float* __restrict__ gx,
                                const __bf16* __restrict__ wsw,
                                const float* __restrict__ wlin,
                                const float* __restrict__ blin,
                                float* __restrict__ out,
                                float* __restrict__ gates,   // 2 * NGATE floats
                                unsigned* __restrict__ bar) {
  __shared__ __align__(32) __bf16 h_bf[HIDDEN];  // 4 KB, shared h (bf16)
  __shared__ float red[WGSIZE];

  const unsigned tid   = threadIdx.x;
  const unsigned lane  = tid & 31u;
  const unsigned wave  = tid >> 5;
  const unsigned wgid  = blockIdx.x;
  const unsigned gwave = wgid * 8u + wave;        // 0..511 -> N-tile
  const unsigned n0    = gwave * 16u;

  // h = 0, c = 0
  #pragma unroll
  for (int i = 0; i < 8; ++i) h_bf[tid + i * WGSIZE] = f2bf(0.0f);
  float cst[8], hst[8];
  #pragma unroll
  for (int i = 0; i < 8; ++i) { cst[i] = 0.0f; hst[i] = 0.0f; }
  __syncthreads();

  const __bf16* wb = wsw + (size_t)gwave * (KTILES * 512);  // 64 KB/wave, contiguous
  const unsigned eo = (lane & 16u);  // element offset 0 or 16 into h chunk

  for (int t = 0; t < SEQ_LEN; ++t) {
    float* gcur = gates + (size_t)(t & 1) * NGATE;

    // ---- h . W_hh.T for this wave's 16 gate rows ----
    v8f acc = {};
    #pragma unroll 4
    for (int kt = 0; kt < KTILES; ++kt) {
      // B fragment: one coalesced 32B/lane load from pre-swizzled weights
      v16bf bfr = *(const v16bf*)(wb + (size_t)kt * 512 + (size_t)lane * 16);
      // A fragment: every row = h chunk (all D rows equal the matvec result)
      v8bf lo = *(const v8bf*)(&h_bf[kt * 32 + eo]);
      v8bf hi = *(const v8bf*)(&h_bf[kt * 32 + 8 + eo]);
      v16bf afr = __builtin_shufflevector(lo, hi,
          0, 1, 2, 3, 4, 5, 6, 7, 8, 9, 10, 11, 12, 13, 14, 15);
      acc = __builtin_amdgcn_wmma_f32_16x16x32_bf16(
          false, afr, false, bfr, (short)0, acc, false, false);
    }
    if (lane < 16u) {
      float g = acc[0] + gx[(size_t)t * NGATE + n0 + lane];
      __hip_atomic_store(&gcur[n0 + lane], g, __ATOMIC_RELAXED,
                         __HIP_MEMORY_SCOPE_AGENT);
    }

    // ---- grid barrier (monotonic counter; double-buffered gates => 1/step) ----
    __threadfence();
    __syncthreads();
    if (tid == 0) {
      __hip_atomic_fetch_add(bar, 1u, __ATOMIC_RELEASE, __HIP_MEMORY_SCOPE_AGENT);
      const unsigned target = (unsigned)(t + 1) * NWG;
      while (__hip_atomic_load(bar, __ATOMIC_ACQUIRE, __HIP_MEMORY_SCOPE_AGENT) < target)
        __builtin_amdgcn_s_sleep(1);
    }
    __syncthreads();

    // ---- elementwise gate update (replicated per WG; c stays in registers) ----
    #pragma unroll
    for (int i = 0; i < 8; ++i) {
      unsigned j = tid + i * WGSIZE;
      float gi = __hip_atomic_load(&gcur[j],              __ATOMIC_RELAXED, __HIP_MEMORY_SCOPE_AGENT);
      float gf = __hip_atomic_load(&gcur[j +     HIDDEN], __ATOMIC_RELAXED, __HIP_MEMORY_SCOPE_AGENT);
      float gg = __hip_atomic_load(&gcur[j + 2 * HIDDEN], __ATOMIC_RELAXED, __HIP_MEMORY_SCOPE_AGENT);
      float go = __hip_atomic_load(&gcur[j + 3 * HIDDEN], __ATOMIC_RELAXED, __HIP_MEMORY_SCOPE_AGENT);
      float si = 1.0f / (1.0f + __expf(-gi));
      float sf = 1.0f / (1.0f + __expf(-gf));
      float tg = tanhf(gg);
      float so = 1.0f / (1.0f + __expf(-go));
      float c  = sf * cst[i] + si * tg;
      float h  = so * tanhf(c);
      cst[i] = c;
      hst[i] = h;
      h_bf[j] = f2bf(h);
    }
    __syncthreads();
  }

  // ---- final head: out = h_last . W_lin + b_lin (WG0 only) ----
  if (wgid == 0) {
    float p = 0.0f;
    #pragma unroll
    for (int i = 0; i < 8; ++i) p += hst[i] * wlin[tid + i * WGSIZE];
    red[tid] = p;
    __syncthreads();
    for (int s = WGSIZE / 2; s > 0; s >>= 1) {
      if (tid < (unsigned)s) red[tid] += red[tid + s];
      __syncthreads();
    }
    if (tid == 0) out[0] = red[0] + blin[0];
  }
}

// ---------------- launcher ----------------
extern "C" void kernel_launch(void* const* d_in, const int* in_sizes, int n_in,
                              void* d_out, int out_size, void* d_ws, size_t ws_size,
                              hipStream_t stream) {
  const float* input_seq = (const float*)d_in[0];
  const float* W_ih      = (const float*)d_in[1];
  const float* W_hh      = (const float*)d_in[2];
  const float* b_ih      = (const float*)d_in[3];
  const float* b_hh      = (const float*)d_in[4];
  const float* W_lin     = (const float*)d_in[5];
  const float* b_lin     = (const float*)d_in[6];
  float* out = (float*)d_out;

  char* ws = (char*)d_ws;
  // workspace layout:
  //   [0,4)        barrier counter
  //   [256, +64KB) double-buffered gates (2 * 8192 f32)
  //   [1MB, +32MB) swizzled bf16 W_hh
  //   [34MB,+128MB) gx (4096 x 8192 f32)
  unsigned* bar   = (unsigned*)(ws + 0);
  float*    gates = (float*)(ws + 256);
  __bf16*   wsw   = (__bf16*)(ws + (1ull << 20));
  float*    gx    = (float*)(ws + (34ull << 20));

  hipLaunchKernelGGL(init_ws, dim3(1), dim3(32), 0, stream, bar);
  hipLaunchKernelGGL(swizzle_whh, dim3((NGATE * HIDDEN) / 256), dim3(256), 0, stream,
                     W_hh, wsw);
  hipLaunchKernelGGL(gx_kernel, dim3(4096), dim3(256), 0, stream,
                     input_seq, W_ih, b_ih, b_hh, gx);
  hipLaunchKernelGGL(lstm_persistent, dim3(NWG), dim3(WGSIZE), 0, stream,
                     gx, wsw, W_lin, b_lin, out, gates, bar);
}
